// ImprovedQuantizedLeNet_16707422782120
// MI455X (gfx1250) — matrix operand order repaired and checked by reading, not compiled, not run on hardware
//
#include <hip/hip_runtime.h>
#include <hip/hip_bf16.h>

typedef __attribute__((ext_vector_type(16))) _Float16 v16h;
typedef __attribute__((ext_vector_type(8)))  float    v8f;
typedef __attribute__((ext_vector_type(4)))  unsigned tdm_v4u;
typedef __attribute__((ext_vector_type(4)))  int      tdm_v4i;
typedef __attribute__((ext_vector_type(8)))  int      tdm_v8i;

#define NBINS 4096
#define C2_KP 160   // conv2 GEMM K: 6*25=150 padded to 5*32

#if __has_builtin(__builtin_amdgcn_tensor_load_to_lds) && \
    __has_builtin(__builtin_amdgcn_s_wait_tensorcnt)
#define HAVE_TDM 1
#endif

// ---------------------------------------------------------------------------
// async memory->LDS staging (CDNA5 GLOBAL_LOAD_ASYNC_TO_LDS, ASYNCcnt)
// lds_off: byte offset in workgroup LDS; ga: 64-bit global byte address
// ---------------------------------------------------------------------------
__device__ __forceinline__ void async_load_lds_b128(unsigned lds_off,
                                                    unsigned long long ga) {
    asm volatile("global_load_async_to_lds_b128 %0, %1, off"
                 :: "v"(lds_off), "v"(ga) : "memory");
}
__device__ __forceinline__ void async_wait_all() {
    asm volatile("s_wait_asynccnt 0x0" ::: "memory");
}

// ---------------------------------------------------------------------------
// quantize-on-read: mirrors reference bit-munging (low nibble -> 0b1001)
// ---------------------------------------------------------------------------
__device__ __forceinline__ float quantize_val(float v, float sc) {
    float qf = floorf(v * sc);
    qf = fminf(fmaxf(qf, -128.0f), 127.0f);
    int q  = (int)qf;
    int u  = q & 0xFF;            // two's-complement byte
    int u2 = (u & 0xF0) | 9;      // low nibble -> 1001b
    int s  = (u2 >= 128) ? (u2 - 256) : u2;
    return (float)s / sc;
}

// ---------------------------------------------------------------------------
// stats init
// ---------------------------------------------------------------------------
__global__ void zero_stats_kernel(unsigned* gmax, float* scaleArr, unsigned* hist) {
    int tid = threadIdx.x;
    if (tid < 4) { gmax[tid] = 0u; scaleArr[tid] = 0.0f; }
    for (int i = tid; i < NBINS; i += blockDim.x) hist[i] = 0u;
}

// pack weights to f16, zero-padded:
//   w1h[128][416]  (fc1), w2h[96][128] (fc2), w2c[16][160] (conv2 implicit GEMM)
__global__ void pack_weights_kernel(const float* __restrict__ w1,
                                    const float* __restrict__ w2,
                                    const float* __restrict__ cw2,
                                    _Float16* __restrict__ w1h,
                                    _Float16* __restrict__ w2h,
                                    _Float16* __restrict__ w2c) {
    const int total1 = 128 * 416, total2 = 96 * 128, total3 = 16 * C2_KP;
    for (int idx = blockIdx.x * blockDim.x + threadIdx.x;
         idx < total1 + total2 + total3; idx += gridDim.x * blockDim.x) {
        if (idx < total1) {
            int n = idx / 416, k = idx % 416;
            float v = (n < 120 && k < 400) ? w1[n * 400 + k] : 0.0f;
            w1h[idx] = (_Float16)v;
        } else if (idx < total1 + total2) {
            int j = idx - total1;
            int n = j / 128, k = j % 128;
            float v = (n < 84 && k < 120) ? w2[n * 120 + k] : 0.0f;
            w2h[j] = (_Float16)v;
        } else {
            int j = idx - total1 - total2;
            int c = j / C2_KP, k = j % C2_KP;     // k = ci*25 + tap
            float v = (k < 150) ? cw2[c * 150 + k] : 0.0f;
            w2c[j] = (_Float16)v;
        }
    }
}

// ---------------------------------------------------------------------------
// conv1 (1->6, 5x5, pad 2) + ReLU + 2x2 maxpool, one sample per block (VALU:
// K=25 single-channel is WMMA-hostile; fragment build cost ~= FMA count).
// Input image staged to LDS via async-to-LDS (196 x B128, ASYNCcnt).
// ---------------------------------------------------------------------------
__global__ void conv1_kernel(const float* __restrict__ x, const float* __restrict__ w,
                             const float* __restrict__ bias, float* __restrict__ a1,
                             unsigned* __restrict__ gmax) {
    __shared__ __align__(16) float xin[784];
    __shared__ float ws[150];
    __shared__ float bs[6];
    __shared__ float red[256];
    const int tid = threadIdx.x, s = blockIdx.x;
    const float* xp = x + (size_t)s * 784;          // 3136 B stride, 16B aligned
    {
        unsigned base = (unsigned)(size_t)&xin[0];
        for (int i = tid; i < 196; i += 256)
            async_load_lds_b128(base + 16u * i,
                                (unsigned long long)(xp + 4 * i));
        async_wait_all();
    }
    if (tid < 150) ws[tid] = w[tid];
    if (tid < 6)   bs[tid] = bias[tid];
    __syncthreads();
    float tmax = 0.0f;
    for (int i = tid; i < 1176; i += 256) {          // 6*14*14 pooled outputs
        int c = i / 196, rem = i % 196, py = rem / 14, px = rem % 14;
        const float* wc = &ws[c * 25];
        float acc = 0.0f;                            // post-ReLU values >= 0
        for (int dy = 0; dy < 2; ++dy)
            for (int dx = 0; dx < 2; ++dx) {
                int oy = 2 * py + dy, ox = 2 * px + dx;
                float sum = bs[c];
#pragma unroll
                for (int ky = 0; ky < 5; ++ky) {
                    int iy = oy + ky - 2;
                    if (iy < 0 || iy >= 28) continue;
#pragma unroll
                    for (int kx = 0; kx < 5; ++kx) {
                        int ix = ox + kx - 2;
                        if (ix >= 0 && ix < 28) sum += xin[iy * 28 + ix] * wc[ky * 5 + kx];
                    }
                }
                acc = fmaxf(acc, fmaxf(sum, 0.0f));
            }
        a1[(size_t)s * 1176 + i] = acc;
        tmax = fmaxf(tmax, acc);
    }
    red[tid] = tmax; __syncthreads();
    for (int st = 128; st > 0; st >>= 1) {
        if (tid < st) red[tid] = fmaxf(red[tid], red[tid + st]);
        __syncthreads();
    }
    if (tid == 0) atomicMax(gmax + 0, __float_as_uint(red[0]));
}

// ---------------------------------------------------------------------------
// histogram over [0, gmax[stage]] -> 4096 bins (LDS-staged)
// ---------------------------------------------------------------------------
__global__ void hist_kernel(const float* __restrict__ src, size_t N,
                            const unsigned* __restrict__ gmax, int stage,
                            unsigned* __restrict__ hist) {
    __shared__ unsigned lh[NBINS];
    for (int i = threadIdx.x; i < NBINS; i += blockDim.x) lh[i] = 0u;
    __syncthreads();
    float mx = __uint_as_float(gmax[stage]);
    float inv = (mx > 0.0f) ? ((float)NBINS / mx) : 0.0f;
    size_t stride = (size_t)gridDim.x * blockDim.x;
    for (size_t i = (size_t)blockIdx.x * blockDim.x + threadIdx.x; i < N; i += stride) {
        int b = (int)(src[i] * inv);
        b = min(max(b, 0), NBINS - 1);
        atomicAdd(&lh[b], 1u);
    }
    __syncthreads();
    for (int i = threadIdx.x; i < NBINS; i += blockDim.x)
        if (lh[i]) atomicAdd(&hist[i], lh[i]);
}

// scan histogram -> percentile-based scale; zero hist for next stage
__global__ void scale_kernel(unsigned* __restrict__ hist,
                             const unsigned* __restrict__ gmax,
                             float* __restrict__ scaleArr,
                             int stage, unsigned long long N) {
    if (threadIdx.x == 0) {
        float mx = __uint_as_float(gmax[stage]);
        float scale;
        if (!(mx > 0.0f)) {
            scale = 128.0f;
        } else {
            float width = mx / (float)NBINS;
            double ranks[2] = {0.025 * (double)(N - 1), 0.975 * (double)(N - 1)};
            float vals[2] = {0.0f, mx};
            unsigned long long cum = 0; int pi = 0;
            for (int b = 0; b < NBINS && pi < 2; ++b) {
                unsigned cnt = hist[b];
                while (pi < 2 && (double)(cum + cnt) > ranks[pi]) {
                    double frac = cnt ? (ranks[pi] - (double)cum) / (double)cnt : 0.0;
                    vals[pi] = ((float)b + (float)frac) * width;
                    ++pi;
                }
                cum += cnt;
            }
            float abs_max = fmaxf(fabsf(vals[0]), fabsf(vals[1]));
            scale = (abs_max > 0.0f) ? (127.0f / abs_max) : 128.0f;
        }
        scaleArr[stage] = scale;
    }
    __syncthreads();
    for (int i = threadIdx.x; i < NBINS; i += blockDim.x) hist[i] = 0u;
}

// ---------------------------------------------------------------------------
// conv2 as implicit GEMM on WMMA: per sample  D[16ch x 112px] = W[16 x 160] *
// im2col[160 x 112];  im2col LDS address = tapOff[k] + pixOff[n].
// Activation block staged into LDS via TDM (if builtin exists) or
// async-to-LDS inline asm (294 x B128, ASYNCcnt).
// ---------------------------------------------------------------------------
__global__ void conv2_wmma_kernel(const float* __restrict__ a1,
                                  const _Float16* __restrict__ w2c, // [16][160]
                                  const float* __restrict__ bias,
                                  const float* __restrict__ scaleArr,
                                  float* __restrict__ a2,
                                  unsigned* __restrict__ gmax) {
    __shared__ __align__(16) float h_raw[1176];
    __shared__ _Float16 hq[1184];
    __shared__ unsigned short tap[C2_KP];
    __shared__ float conv_out[16][102];   // stride 102 breaks bank alignment
    __shared__ float red[256];
    const int tid = threadIdx.x, s = blockIdx.x;
    const float sc = scaleArr[0];
    const float* hp = a1 + (size_t)s * 1176;        // 4704 B stride, 16B aligned

#ifdef HAVE_TDM
    if (tid < 32) {   // wave 0 issues one TDM descriptor for the sample block
        unsigned ldsa = (unsigned)(unsigned long long)&h_raw[0];
        unsigned long long ga = (unsigned long long)hp;
        tdm_v4u g0; tdm_v4i gz; tdm_v8i g1;
        g0[0] = 1u;                                   // count=1 (valid D#)
        g0[1] = ldsa;                                 // lds_addr
        g0[2] = (unsigned)ga;                         // global_addr[31:0]
        g0[3] = (unsigned)(ga >> 32) | (2u << 30);    // global_addr[56:32], type=2
        g1[0] = (int)0x00020000u;                     // data_size=4B, no mask
        g1[1] = (int)(1176u << 16);                   // tensor_dim0 = 1176
        g1[2] = (int)0x00010000u;                     // tensor_dim1 = 1
        g1[3] = (int)(1176u << 16);                   // tile_dim0 = 1176
        g1[4] = 1;                                    // tile_dim1 = 1
        g1[5] = 1176;                                 // tensor_dim0_stride
        g1[6] = 0; g1[7] = 0;
        gz[0] = gz[1] = gz[2] = gz[3] = 0;
#if defined(__clang_major__) && (__clang_major__ >= 23)
        tdm_v8i gz8; gz8[0]=gz8[1]=gz8[2]=gz8[3]=gz8[4]=gz8[5]=gz8[6]=gz8[7]=0;
        __builtin_amdgcn_tensor_load_to_lds(g0, g1, gz, gz, gz8, 0);
#else
        __builtin_amdgcn_tensor_load_to_lds(g0, g1, gz, gz, 0);
#endif
        __builtin_amdgcn_s_wait_tensorcnt(0);
    }
#else
    {
        unsigned base = (unsigned)(size_t)&h_raw[0];
        for (int i = tid; i < 294; i += 256)          // 1176 f32 = 294 x 16B
            async_load_lds_b128(base + 16u * i,
                                (unsigned long long)(hp + 4 * i));
        async_wait_all();
    }
#endif
    if (tid < C2_KP) {                 // tapOff[k]: k = ci*25 + ky*5 + kx
        unsigned short off = 0;
        if (tid < 150) {
            int ci = tid / 25, t2 = tid % 25;
            off = (unsigned short)(ci * 196 + (t2 / 5) * 14 + (t2 % 5));
        }
        tap[tid] = off;                // k >= 150: A is zero-padded, B value moot
    }
    __syncthreads();
    for (int i = tid; i < 1176; i += 256) hq[i] = (_Float16)quantize_val(h_raw[i], sc);
    __syncthreads();

    const int wave = tid >> 5, lane = tid & 31;
    if (wave < 7) {                    // 7 N-tiles cover 112 cols (100 valid)
        const int row = lane & 15, hi = lane >> 4;
        const int nB = wave * 16 + row;
        const int pixOff = (nB < 100) ? ((nB / 10) * 14 + (nB % 10)) : 0;
        v8f c = {};
        for (int ks = 0; ks < 5; ++ks) {
            const int kb = ks * 32;
            v16h af, bf;
            const _Float16* wp = w2c + row * C2_KP + kb;   // A = weights
#pragma unroll
            for (int j = 0; j < 8; ++j)  af[j] = wp[hi * 16 + j];
#pragma unroll
            for (int j = 8; j < 16; ++j) af[j] = wp[hi * 16 + j];
            const int kbb = kb + hi * 16;
#pragma unroll
            for (int j = 0; j < 16; ++j) bf[j] = hq[tap[kbb + j] + pixOff];
            c = __builtin_amdgcn_wmma_f32_16x16x32_f16(false, af, false, bf,
                                                       (short)0, c, false, false);
        }
        const int nn = wave * 16 + (lane & 15);
        if (nn < 100) {
#pragma unroll
            for (int r = 0; r < 8; ++r) {
                int ch = r + hi * 8;
                conv_out[ch][nn] = fmaxf(c[r] + bias[ch], 0.0f);
            }
        }
    }
    __syncthreads();
    float tmax = 0.0f;                 // 2x2 maxpool -> a2 (400 per sample)
    for (int t = tid; t < 400; t += 256) {
        int ch = t / 25, rem = t % 25, py = rem / 5, px = rem % 5;
        int n00 = (2 * py) * 10 + 2 * px;
        float v = fmaxf(fmaxf(conv_out[ch][n00],      conv_out[ch][n00 + 1]),
                        fmaxf(conv_out[ch][n00 + 10], conv_out[ch][n00 + 11]));
        a2[(size_t)s * 400 + t] = v;
        tmax = fmaxf(tmax, v);
    }
    red[tid] = tmax; __syncthreads();
    for (int st = 128; st > 0; st >>= 1) {
        if (tid < st) red[tid] = fmaxf(red[tid], red[tid + st]);
        __syncthreads();
    }
    if (tid == 0) atomicMax(gmax + 1, __float_as_uint(red[0]));
}

// ---------------------------------------------------------------------------
// FC1: [B,400] x [400,120]^T. Block = 256 thr = 8 waves computing M=64,N=128;
// each wave: 1 B-fragment reused by 4 A-tiles (4 accumulators), K=13*32.
// ---------------------------------------------------------------------------
__global__ void fc1_wmma_kernel(const float* __restrict__ a2,
                                const _Float16* __restrict__ w1h,   // [128][416]
                                const float* __restrict__ bias,
                                const float* __restrict__ scaleArr,
                                float* __restrict__ f1,
                                unsigned* __restrict__ gmax) {
    __shared__ _Float16 At[64][424];     // 54 KB, stride 424 -> conflict-free
    __shared__ float red[256];
    const int tid = threadIdx.x;
    const int m0  = blockIdx.x * 64;
    const float sc = scaleArr[1];
    for (int idx = tid; idx < 64 * 416; idx += 256) {
        int m = idx / 416, k = idx % 416;
        float v = (k < 400) ? quantize_val(a2[(size_t)(m0 + m) * 400 + k], sc) : 0.0f;
        At[m][k] = (_Float16)v;
    }
    __syncthreads();
    const int wave = tid >> 5, lane = tid & 31;
    const int row = lane & 15, hi = lane >> 4;
    const int n0 = wave * 16;
    v8f cacc[4];
#pragma unroll
    for (int mt = 0; mt < 4; ++mt) cacc[mt] = (v8f){0,0,0,0,0,0,0,0};
    for (int ks = 0; ks < 13; ++ks) {
        const int kb = ks * 32;
        v16h bf;
        const _Float16* wp = w1h + (size_t)(n0 + row) * 416 + kb + hi * 16;
#pragma unroll
        for (int j = 0; j < 16; ++j) bf[j] = wp[j];
#pragma unroll
        for (int mt = 0; mt < 4; ++mt) {
            v16h af;
            const int mr = mt * 16 + row;
#pragma unroll
            for (int j = 0; j < 8; ++j)  af[j] = At[mr][kb + hi * 16 + j];
#pragma unroll
            for (int j = 8; j < 16; ++j) af[j] = At[mr][kb + hi * 16 + j];
            cacc[mt] = __builtin_amdgcn_wmma_f32_16x16x32_f16(false, af, false, bf,
                                                              (short)0, cacc[mt],
                                                              false, false);
        }
    }
    float tmax = 0.0f;
    const int nn = n0 + (lane & 15);
#pragma unroll
    for (int mt = 0; mt < 4; ++mt) {
#pragma unroll
        for (int r = 0; r < 8; ++r) {
            if (nn < 120) {
                int m = m0 + mt * 16 + r + hi * 8;
                float v = fmaxf(cacc[mt][r] + bias[nn], 0.0f);
                f1[(size_t)m * 120 + nn] = v;
                tmax = fmaxf(tmax, v);
            }
        }
    }
    red[tid] = tmax; __syncthreads();
    for (int st = 128; st > 0; st >>= 1) {
        if (tid < st) red[tid] = fmaxf(red[tid], red[tid + st]);
        __syncthreads();
    }
    if (tid == 0) atomicMax(gmax + 2, __float_as_uint(red[0]));
}

// ---------------------------------------------------------------------------
// FC2: [B,120] x [120,84]^T. Block = 192 thr = 6 waves, M=64, N=96, K=128.
// ---------------------------------------------------------------------------
__global__ void fc2_wmma_kernel(const float* __restrict__ f1,
                                const _Float16* __restrict__ w2h,   // [96][128]
                                const float* __restrict__ bias,
                                const float* __restrict__ scaleArr,
                                float* __restrict__ f2,
                                unsigned* __restrict__ gmax) {
    __shared__ _Float16 At[64][136];
    __shared__ float red[256];
    const int tid = threadIdx.x;
    const int m0  = blockIdx.x * 64;
    const float sc = scaleArr[2];
    for (int idx = tid; idx < 64 * 128; idx += 192) {
        int m = idx >> 7, k = idx & 127;
        float v = (k < 120) ? quantize_val(f1[(size_t)(m0 + m) * 120 + k], sc) : 0.0f;
        At[m][k] = (_Float16)v;
    }
    __syncthreads();
    const int wave = tid / 32, lane = tid & 31;
    const int row = lane & 15, hi = lane >> 4;
    const int n0 = wave * 16;
    v8f cacc[4];
#pragma unroll
    for (int mt = 0; mt < 4; ++mt) cacc[mt] = (v8f){0,0,0,0,0,0,0,0};
    for (int ks = 0; ks < 4; ++ks) {
        const int kb = ks * 32;
        v16h bf;
        const _Float16* wp = w2h + (size_t)(n0 + row) * 128 + kb + hi * 16;
#pragma unroll
        for (int j = 0; j < 16; ++j) bf[j] = wp[j];
#pragma unroll
        for (int mt = 0; mt < 4; ++mt) {
            v16h af;
            const int mr = mt * 16 + row;
#pragma unroll
            for (int j = 0; j < 8; ++j)  af[j] = At[mr][kb + hi * 16 + j];
#pragma unroll
            for (int j = 8; j < 16; ++j) af[j] = At[mr][kb + hi * 16 + j];
            cacc[mt] = __builtin_amdgcn_wmma_f32_16x16x32_f16(false, af, false, bf,
                                                              (short)0, cacc[mt],
                                                              false, false);
        }
    }
    float tmax = 0.0f;
    const int nn = n0 + (lane & 15);
#pragma unroll
    for (int mt = 0; mt < 4; ++mt) {
#pragma unroll
        for (int r = 0; r < 8; ++r) {
            if (nn < 84) {
                int m = m0 + mt * 16 + r + hi * 8;
                float v = fmaxf(cacc[mt][r] + bias[nn], 0.0f);
                f2[(size_t)m * 84 + nn] = v;
                tmax = fmaxf(tmax, v);
            }
        }
    }
    red[tid] = tmax;
    if (tid < 64) red[192 + tid] = 0.0f;
    __syncthreads();
    for (int st = 128; st > 0; st >>= 1) {
        if (tid < st) red[tid] = fmaxf(red[tid], red[tid + st]);
        __syncthreads();
    }
    if (tid == 0) atomicMax(gmax + 3, __float_as_uint(red[0]));
}

// ---------------------------------------------------------------------------
// FC3: [B,84] x [84,10]^T, quantize-on-read; tiny (27 MFLOP) -> VALU
// ---------------------------------------------------------------------------
__global__ void fc3_kernel(const float* __restrict__ f2, const float* __restrict__ w,
                           const float* __restrict__ bias,
                           const float* __restrict__ scaleArr,
                           float* __restrict__ out, int B) {
    int idx = blockIdx.x * blockDim.x + threadIdx.x;
    if (idx >= B * 10) return;
    int m = idx / 10, n = idx % 10;
    float sc = scaleArr[3];
    float acc = bias[n];
    const float* fp = f2 + (size_t)m * 84;
    const float* wp = w + n * 84;
#pragma unroll 4
    for (int k = 0; k < 84; ++k) acc += quantize_val(fp[k], sc) * wp[k];
    out[idx] = acc;
}

// ---------------------------------------------------------------------------
extern "C" void kernel_launch(void* const* d_in, const int* in_sizes, int n_in,
                              void* d_out, int out_size, void* d_ws, size_t ws_size,
                              hipStream_t stream) {
    const float* x   = (const float*)d_in[0];
    const float* cw1 = (const float*)d_in[1];
    const float* cb1 = (const float*)d_in[2];
    const float* cw2 = (const float*)d_in[3];
    const float* cb2 = (const float*)d_in[4];
    const float* w1  = (const float*)d_in[5];
    const float* b1  = (const float*)d_in[6];
    const float* w2  = (const float*)d_in[7];
    const float* b2  = (const float*)d_in[8];
    const float* w3  = (const float*)d_in[9];
    const float* b3  = (const float*)d_in[10];
    float* out = (float*)d_out;

    const int B = in_sizes[0] / (28 * 28);   // 16384

    char* ws = (char*)d_ws;
    size_t cur = 0;
    auto take = [&](size_t bytes) -> char* {
        size_t off = (cur + 255) & ~(size_t)255;
        cur = off + bytes;
        return ws + off;
    };
    unsigned* gmax     = (unsigned*)take(4 * sizeof(unsigned));
    float*    scaleArr = (float*)   take(4 * sizeof(float));
    unsigned* hist     = (unsigned*)take(NBINS * sizeof(unsigned));
    _Float16* w1h      = (_Float16*)take((size_t)128 * 416 * sizeof(_Float16));
    _Float16* w2h      = (_Float16*)take((size_t)96 * 128 * sizeof(_Float16));
    _Float16* w2c      = (_Float16*)take((size_t)16 * C2_KP * sizeof(_Float16));
    float*    a1       = (float*)   take((size_t)B * 1176 * sizeof(float));
    float*    a2       = (float*)   take((size_t)B * 400  * sizeof(float));
    float*    f1       = (float*)   take((size_t)B * 120  * sizeof(float));
    float*    f2       = (float*)   take((size_t)B * 84   * sizeof(float));

    zero_stats_kernel<<<1, 256, 0, stream>>>(gmax, scaleArr, hist);
    pack_weights_kernel<<<128, 256, 0, stream>>>(w1, w2, cw2, w1h, w2h, w2c);

    // stage 0: conv1 + relu + pool -> a1 ; percentile -> scale[0]
    conv1_kernel<<<B, 256, 0, stream>>>(x, cw1, cb1, a1, gmax);
    hist_kernel<<<2048, 256, 0, stream>>>(a1, (size_t)B * 1176, gmax, 0, hist);
    scale_kernel<<<1, 256, 0, stream>>>(hist, gmax, scaleArr, 0,
                                        (unsigned long long)B * 1176ull);

    // stage 1: conv2 implicit-GEMM WMMA -> a2 ; scale[1]
    conv2_wmma_kernel<<<B, 256, 0, stream>>>(a1, w2c, cb2, scaleArr, a2, gmax);
    hist_kernel<<<2048, 256, 0, stream>>>(a2, (size_t)B * 400, gmax, 1, hist);
    scale_kernel<<<1, 256, 0, stream>>>(hist, gmax, scaleArr, 1,
                                        (unsigned long long)B * 400ull);

    // stage 2: fc1 WMMA (M=64/block) -> f1 ; scale[2]
    fc1_wmma_kernel<<<B / 64, 256, 0, stream>>>(a2, w1h, b1, scaleArr, f1, gmax);
    hist_kernel<<<1024, 256, 0, stream>>>(f1, (size_t)B * 120, gmax, 2, hist);
    scale_kernel<<<1, 256, 0, stream>>>(hist, gmax, scaleArr, 2,
                                        (unsigned long long)B * 120ull);

    // stage 3: fc2 WMMA (M=64/block) -> f2 ; scale[3]
    fc2_wmma_kernel<<<B / 64, 192, 0, stream>>>(f1, w2h, b2, scaleArr, f2, gmax);
    hist_kernel<<<1024, 256, 0, stream>>>(f2, (size_t)B * 84, gmax, 3, hist);
    scale_kernel<<<1, 256, 0, stream>>>(hist, gmax, scaleArr, 3,
                                        (unsigned long long)B * 84ull);

    // fc3 -> output
    fc3_kernel<<<(B * 10 + 255) / 256, 256, 0, stream>>>(f2, w3, b3, scaleArr, out, B);
    (void)n_in; (void)out_size; (void)ws_size;
}
